// GptOssMLP_75557064671537
// MI455X (gfx1250) — matrix operand order repaired and checked by reading, not compiled, not run on hardware
//
#include <hip/hip_runtime.h>

#define HD   2880     // hidden size
#define ID   2880     // intermediate size
#define NE   8        // experts
#define NT   2048     // tokens
#define BM   128
#define BN   64
#define KT   32
#define PADROWS 5120
#define MAXTILES 40
#define GALPHA 1.702f
#define GLIMIT 7.0f

typedef __bf16 bf16;
typedef __attribute__((ext_vector_type(8)))  bf16  v8bf;
typedef __attribute__((ext_vector_type(16))) bf16  v16bf;
typedef __attribute__((ext_vector_type(8)))  float v8f;
typedef __attribute__((ext_vector_type(4)))  int   v4i;

#if __has_builtin(__builtin_amdgcn_global_load_async_to_lds_b128)
#define HAVE_ASYNC 1
#else
#define HAVE_ASYNC 0
#endif

#if HAVE_ASYNC
typedef __attribute__((address_space(1))) v4i* as1_v4i;
typedef __attribute__((address_space(3))) v4i* as3_v4i;
__device__ __forceinline__ void cp_async16(void* lds, const void* g) {
    __builtin_amdgcn_global_load_async_to_lds_b128((as1_v4i)(g), (as3_v4i)(lds), 0, 0);
}
#if __has_builtin(__builtin_amdgcn_s_wait_asynccnt)
#define WAIT_ASYNC() __builtin_amdgcn_s_wait_asynccnt(0)
#else
#define WAIT_ASYNC() asm volatile("s_wait_asynccnt 0x0" ::: "memory")
#endif
#endif

// ---------------- init: zero atomic counters ----------------
__global__ void k_init(int* counts, int* cursor) {
    int i = threadIdx.x;
    if (i < NE) { counts[i] = 0; cursor[i] = 0; }
}

// ---- convert gate_up [E][HD][2*ID] f32 -> Wg[E][ID][HD], Wu[E][ID][HD] bf16
//      (transpose to K-major + de-interleave gate/up) ----
__global__ __launch_bounds__(256) void k_conv_gateup(const float* __restrict__ gup,
                                                     bf16* __restrict__ Wg,
                                                     bf16* __restrict__ Wu) {
    int bid = blockIdx.x;
    int hT = bid % 45; int nT = (bid / 45) % 45; int e = bid / (45 * 45);
    int h0 = hT * 64, n0 = nT * 64;
    __shared__ float tile[64][132];            // [h][interleaved col], padded
    int tid = threadIdx.x;
    const float* src = gup + (size_t)e * HD * (2 * ID);
    #pragma unroll
    for (int j = 0; j < 8; ++j) {
        int idx = tid + j * 256;               // 2048 float4 slots
        int r  = idx >> 5;                     // 64 rows (h)
        int c4 = (idx & 31) * 4;               // 128 floats per row
        float4 v = *(const float4*)&src[(size_t)(h0 + r) * (2 * ID) + 2 * n0 + c4];
        tile[r][c4 + 0] = v.x; tile[r][c4 + 1] = v.y;
        tile[r][c4 + 2] = v.z; tile[r][c4 + 3] = v.w;
    }
    __syncthreads();
    int n = tid >> 2;                          // 0..63 (intermediate col)
    int hseg = (tid & 3) * 16;                 // 0,16,32,48
    v8bf g0, g1, u0, u1;
    #pragma unroll
    for (int i = 0; i < 8; ++i) {
        g0[i] = (bf16)tile[hseg + i][2 * n];
        u0[i] = (bf16)tile[hseg + i][2 * n + 1];
        g1[i] = (bf16)tile[hseg + 8 + i][2 * n];
        u1[i] = (bf16)tile[hseg + 8 + i][2 * n + 1];
    }
    size_t o = ((size_t)e * ID + (n0 + n)) * HD + h0 + hseg;
    *(v8bf*)&Wg[o] = g0; *(v8bf*)&Wg[o + 8] = g1;
    *(v8bf*)&Wu[o] = u0; *(v8bf*)&Wu[o + 8] = u1;
}

// ---- convert down_proj [E][ID][HD] f32 -> Wd[E][HD][ID] bf16 (K-major) ----
__global__ __launch_bounds__(256) void k_conv_down(const float* __restrict__ dpw,
                                                   bf16* __restrict__ Wd) {
    int bid = blockIdx.x;
    int kT = bid % 45; int nT = (bid / 45) % 45; int e = bid / (45 * 45);
    int k0 = kT * 64, n0 = nT * 64;
    __shared__ float tile[64][68];             // [k][n], padded
    int tid = threadIdx.x;
    const float* src = dpw + (size_t)e * ID * HD;
    #pragma unroll
    for (int j = 0; j < 4; ++j) {
        int idx = tid + j * 256;               // 1024 float4 slots
        int r  = idx >> 4;                     // 64 rows (k)
        int c4 = (idx & 15) * 4;               // 64 floats per row
        float4 v = *(const float4*)&src[(size_t)(k0 + r) * HD + n0 + c4];
        tile[r][c4 + 0] = v.x; tile[r][c4 + 1] = v.y;
        tile[r][c4 + 2] = v.z; tile[r][c4 + 3] = v.w;
    }
    __syncthreads();
    int n = tid >> 2;
    int kseg = (tid & 3) * 16;
    v8bf a, b;
    #pragma unroll
    for (int i = 0; i < 8; ++i) {
        a[i] = (bf16)tile[kseg + i][n];
        b[i] = (bf16)tile[kseg + 8 + i][n];
    }
    size_t o = ((size_t)e * HD + (n0 + n)) * ID + k0 + kseg;
    *(v8bf*)&Wd[o] = a; *(v8bf*)&Wd[o + 8] = b;
}

// ---------------- router: logits -> softmax -> top2 -> renorm ----------------
__global__ __launch_bounds__(256) void k_router(const float* __restrict__ x,
                                                const float* __restrict__ rw,
                                                const float* __restrict__ rb,
                                                int* __restrict__ topk_i,
                                                float* __restrict__ topk_w,
                                                int* __restrict__ counts) {
    int wid = threadIdx.x >> 5, lane = threadIdx.x & 31;
    int t = blockIdx.x * 8 + wid;
    float acc[NE];
    #pragma unroll
    for (int e = 0; e < NE; ++e) acc[e] = 0.f;
    const float* xr = x + (size_t)t * HD;
    for (int h = lane; h < HD; h += 32) {
        float xv = xr[h];
        const float* w = rw + (size_t)h * NE;
        #pragma unroll
        for (int e = 0; e < NE; ++e) acc[e] += xv * w[e];
    }
    #pragma unroll
    for (int off = 16; off > 0; off >>= 1) {
        #pragma unroll
        for (int e = 0; e < NE; ++e) acc[e] += __shfl_xor(acc[e], off, 32);
    }
    if (lane == 0) {
        float m = -1e30f;
        #pragma unroll
        for (int e = 0; e < NE; ++e) { acc[e] += rb[e]; m = fmaxf(m, acc[e]); }
        float p[NE]; float s = 0.f;
        #pragma unroll
        for (int e = 0; e < NE; ++e) { p[e] = __expf(acc[e] - m); s += p[e]; }
        float inv = 1.f / s;
        #pragma unroll
        for (int e = 0; e < NE; ++e) p[e] *= inv;
        int i0 = 0; float m0 = p[0];
        #pragma unroll
        for (int e = 1; e < NE; ++e) if (p[e] > m0) { m0 = p[e]; i0 = e; }
        int i1 = (i0 == 0) ? 1 : 0; float m1 = p[i1];
        #pragma unroll
        for (int e = 0; e < NE; ++e) if (e != i0 && p[e] > m1) { m1 = p[e]; i1 = e; }
        float denom = m0 + m1 + 1e-20f;
        topk_i[t * 2] = i0; topk_i[t * 2 + 1] = i1;
        topk_w[t * 2] = m0 / denom; topk_w[t * 2 + 1] = m1 / denom;
        atomicAdd(&counts[i0], 1); atomicAdd(&counts[i1], 1);
    }
}

// ---------------- tile table (single thread) ----------------
__global__ void k_tiles(const int* __restrict__ counts, int* __restrict__ rowBase,
                        int* __restrict__ meta, int* __restrict__ tileE,
                        int* __restrict__ tileRow) {
    if (threadIdx.x == 0) {
        int base = 0, nt = 0;
        for (int e = 0; e < NE; ++e) {
            rowBase[e] = base;
            int c = counts[e];
            int tc = (c + BM - 1) / BM;
            for (int m = 0; m < tc; ++m) { tileE[nt] = e; tileRow[nt] = base + m * BM; ++nt; }
            base += tc * BM;
        }
        meta[0] = nt;
    }
}

// ---------------- scatter: assign rows, copy x -> bf16 sorted rows ----------------
__global__ __launch_bounds__(256) void k_scatter(const float* __restrict__ x,
                                                 const int* __restrict__ topk_i,
                                                 const int* __restrict__ rowBase,
                                                 int* __restrict__ cursor,
                                                 int* __restrict__ pair_row,
                                                 bf16* __restrict__ Xs) {
    int pair = blockIdx.x; int t = pair >> 1;
    __shared__ int srow;
    if (threadIdx.x == 0) {
        int e = topk_i[pair];
        int pos = atomicAdd(&cursor[e], 1);
        int row = rowBase[e] + pos;
        pair_row[pair] = row;
        srow = row;
    }
    __syncthreads();
    int row = srow;
    const float* xr = x + (size_t)t * HD;
    bf16* dst = Xs + (size_t)row * HD;
    for (int h = threadIdx.x; h < HD; h += 256) dst[h] = (bf16)xr[h];
}

// ---------------- GEMM1: [rows x HD] @ Wg/Wu -> GLU -> Act bf16 ----------------
__global__ __launch_bounds__(256) void k_gemm1(const bf16* __restrict__ Xs,
                                               const bf16* __restrict__ Wg,
                                               const bf16* __restrict__ Wu,
                                               const float* __restrict__ gub,
                                               bf16* __restrict__ Act,
                                               const int* __restrict__ meta,
                                               const int* __restrict__ tileE,
                                               const int* __restrict__ tileRow) {
    int tIdx = blockIdx.y;
    if (tIdx >= meta[0]) return;
    int e = tileE[tIdx]; int mBase = tileRow[tIdx];
    int n0 = blockIdx.x * BN;
    __shared__ __attribute__((aligned(64))) bf16 As[2][BM * KT];
    __shared__ __attribute__((aligned(64))) bf16 Bgs[2][BN * KT];
    __shared__ __attribute__((aligned(64))) bf16 Bus[2][BN * KT];
    int tid = threadIdx.x, lane = tid & 31, wid = tid >> 5;
    int wm = wid & 3, wn = wid >> 2;                // 4x2 wave grid, 32x32 per wave
    v8f accG[2][2], accU[2][2];
    #pragma unroll
    for (int a = 0; a < 2; ++a)
        #pragma unroll
        for (int b = 0; b < 2; ++b)
            #pragma unroll
            for (int r = 0; r < 8; ++r) { accG[a][b][r] = 0.f; accU[a][b][r] = 0.f; }

    const bf16* wgE = Wg + (size_t)e * ID * HD;
    const bf16* wuE = Wu + (size_t)e * ID * HD;
    // per-thread copy slots: A covers 128 rows (two 64-row halves), B covers 64 rows
    int rA = tid >> 2, c8 = (tid & 3) * 8;
    const bf16* gA0 = Xs + (size_t)(mBase + rA) * HD + c8;
    const bf16* gA1 = gA0 + (size_t)64 * HD;
    const bf16* gBg = wgE + (size_t)(n0 + rA) * HD + c8;
    const bf16* gBu = wuE + (size_t)(n0 + rA) * HD + c8;
    unsigned lA0 = rA * KT + c8;
    unsigned lA1 = lA0 + 64 * KT;
    unsigned lB  = rA * KT + c8;
    int kb   = (lane & 16) ? 16 : 0;                // A operand K split
    int koff = (lane & 16) ? 16 : 0;                // B operand K split

    // -------- prologue: fill buffer 0 with k-tile 0 --------
#if HAVE_ASYNC
    cp_async16(&As[0][lA0], gA0);
    cp_async16(&As[0][lA1], gA1);
    cp_async16(&Bgs[0][lB], gBg);
    cp_async16(&Bus[0][lB], gBu);
    WAIT_ASYNC();
#else
    *(v8bf*)&As[0][lA0] = *(const v8bf*)gA0;
    *(v8bf*)&As[0][lA1] = *(const v8bf*)gA1;
    *(v8bf*)&Bgs[0][lB] = *(const v8bf*)gBg;
    *(v8bf*)&Bus[0][lB] = *(const v8bf*)gBu;
#endif
    __syncthreads();

    const int NK = HD / KT;
    for (int i = 0; i < NK; ++i) {
        int cur = i & 1, nxt = cur ^ 1;
        bool more = (i + 1) < NK;
        int kn = (i + 1) * KT;
#if HAVE_ASYNC
        if (more) {
            cp_async16(&As[nxt][lA0], gA0 + kn);
            cp_async16(&As[nxt][lA1], gA1 + kn);
            cp_async16(&Bgs[nxt][lB], gBg + kn);
            cp_async16(&Bus[nxt][lB], gBu + kn);
        }
#else
        v8bf sA0, sA1, sBg, sBu;
        if (more) {
            sA0 = *(const v8bf*)(gA0 + kn);
            sA1 = *(const v8bf*)(gA1 + kn);
            sBg = *(const v8bf*)(gBg + kn);
            sBu = *(const v8bf*)(gBu + kn);
        }
#endif
        v16bf aF[2], gF[2], uF[2];
        #pragma unroll
        for (int s = 0; s < 2; ++s) {
            int m = wm * 32 + s * 16 + (lane & 15);
            v8bf lo = *(const v8bf*)&As[cur][m * KT + kb];
            v8bf hi = *(const v8bf*)&As[cur][m * KT + kb + 8];
            aF[s] = __builtin_shufflevector(lo, hi, 0,1,2,3,4,5,6,7,8,9,10,11,12,13,14,15);
            int n = wn * 32 + s * 16 + (lane & 15);
            gF[s] = *(const v16bf*)&Bgs[cur][n * KT + koff];
            uF[s] = *(const v16bf*)&Bus[cur][n * KT + koff];
        }
        #pragma unroll
        for (int sm = 0; sm < 2; ++sm)
            #pragma unroll
            for (int sn = 0; sn < 2; ++sn) {
                accG[sm][sn] = __builtin_amdgcn_wmma_f32_16x16x32_bf16(
                    false, aF[sm], false, gF[sn], (short)0, accG[sm][sn], false, false);
                accU[sm][sn] = __builtin_amdgcn_wmma_f32_16x16x32_bf16(
                    false, aF[sm], false, uF[sn], (short)0, accU[sm][sn], false, false);
            }
#if HAVE_ASYNC
        if (more) WAIT_ASYNC();
#else
        if (more) {
            *(v8bf*)&As[nxt][lA0] = sA0;
            *(v8bf*)&As[nxt][lA1] = sA1;
            *(v8bf*)&Bgs[nxt][lB] = sBg;
            *(v8bf*)&Bus[nxt][lB] = sBu;
        }
#endif
        __syncthreads();
    }
    // epilogue: bias + clamp + sigmoid-GLU, store bf16 activations
    #pragma unroll
    for (int sm = 0; sm < 2; ++sm) {
        int rowB = mBase + wm * 32 + sm * 16 + ((lane >> 4) << 3);
        #pragma unroll
        for (int sn = 0; sn < 2; ++sn) {
            int j = n0 + wn * 32 + sn * 16 + (lane & 15);
            float2 bias = *(const float2*)&gub[(size_t)e * (2 * ID) + 2 * j];
            #pragma unroll
            for (int r = 0; r < 8; ++r) {
                float g = accG[sm][sn][r] + bias.x;
                float u = accU[sm][sn][r] + bias.y;
                g = fminf(g, GLIMIT);
                u = fminf(fmaxf(u, -GLIMIT), GLIMIT);
                float glu = g * (1.0f / (1.0f + __expf(-GALPHA * g)));
                Act[(size_t)(rowB + r) * ID + j] = (bf16)((u + 1.0f) * glu);
            }
        }
    }
}

// ---------------- GEMM2: Act @ Wd + bias -> Y f32 ----------------
__global__ __launch_bounds__(256) void k_gemm2(const bf16* __restrict__ Act,
                                               const bf16* __restrict__ Wd,
                                               const float* __restrict__ dpb,
                                               float* __restrict__ Y,
                                               const int* __restrict__ meta,
                                               const int* __restrict__ tileE,
                                               const int* __restrict__ tileRow) {
    int tIdx = blockIdx.y;
    if (tIdx >= meta[0]) return;
    int e = tileE[tIdx]; int mBase = tileRow[tIdx];
    int n0 = blockIdx.x * BN;
    __shared__ __attribute__((aligned(64))) bf16 As[2][BM * KT];
    __shared__ __attribute__((aligned(64))) bf16 Bs[2][BN * KT];
    int tid = threadIdx.x, lane = tid & 31, wid = tid >> 5;
    int wm = wid & 3, wn = wid >> 2;
    v8f acc[2][2];
    #pragma unroll
    for (int a = 0; a < 2; ++a)
        #pragma unroll
        for (int b = 0; b < 2; ++b)
            #pragma unroll
            for (int r = 0; r < 8; ++r) acc[a][b][r] = 0.f;

    const bf16* wdE = Wd + (size_t)e * HD * ID;     // [n(HD)][k(ID)]
    int rA = tid >> 2, c8 = (tid & 3) * 8;
    const bf16* gA0 = Act + (size_t)(mBase + rA) * ID + c8;
    const bf16* gA1 = gA0 + (size_t)64 * ID;
    const bf16* gB  = wdE + (size_t)(n0 + rA) * ID + c8;
    unsigned lA0 = rA * KT + c8;
    unsigned lA1 = lA0 + 64 * KT;
    unsigned lB  = rA * KT + c8;
    int kb   = (lane & 16) ? 16 : 0;
    int koff = (lane & 16) ? 16 : 0;

#if HAVE_ASYNC
    cp_async16(&As[0][lA0], gA0);
    cp_async16(&As[0][lA1], gA1);
    cp_async16(&Bs[0][lB], gB);
    WAIT_ASYNC();
#else
    *(v8bf*)&As[0][lA0] = *(const v8bf*)gA0;
    *(v8bf*)&As[0][lA1] = *(const v8bf*)gA1;
    *(v8bf*)&Bs[0][lB]  = *(const v8bf*)gB;
#endif
    __syncthreads();

    const int NK = ID / KT;
    for (int i = 0; i < NK; ++i) {
        int cur = i & 1, nxt = cur ^ 1;
        bool more = (i + 1) < NK;
        int kn = (i + 1) * KT;
#if HAVE_ASYNC
        if (more) {
            cp_async16(&As[nxt][lA0], gA0 + kn);
            cp_async16(&As[nxt][lA1], gA1 + kn);
            cp_async16(&Bs[nxt][lB], gB + kn);
        }
#else
        v8bf sA0, sA1, sB;
        if (more) {
            sA0 = *(const v8bf*)(gA0 + kn);
            sA1 = *(const v8bf*)(gA1 + kn);
            sB  = *(const v8bf*)(gB + kn);
        }
#endif
        v16bf aF[2], bF[2];
        #pragma unroll
        for (int s = 0; s < 2; ++s) {
            int m = wm * 32 + s * 16 + (lane & 15);
            v8bf lo = *(const v8bf*)&As[cur][m * KT + kb];
            v8bf hi = *(const v8bf*)&As[cur][m * KT + kb + 8];
            aF[s] = __builtin_shufflevector(lo, hi, 0,1,2,3,4,5,6,7,8,9,10,11,12,13,14,15);
            int n = wn * 32 + s * 16 + (lane & 15);
            bF[s] = *(const v16bf*)&Bs[cur][n * KT + koff];
        }
        #pragma unroll
        for (int sm = 0; sm < 2; ++sm)
            #pragma unroll
            for (int sn = 0; sn < 2; ++sn)
                acc[sm][sn] = __builtin_amdgcn_wmma_f32_16x16x32_bf16(
                    false, aF[sm], false, bF[sn], (short)0, acc[sm][sn], false, false);
#if HAVE_ASYNC
        if (more) WAIT_ASYNC();
#else
        if (more) {
            *(v8bf*)&As[nxt][lA0] = sA0;
            *(v8bf*)&As[nxt][lA1] = sA1;
            *(v8bf*)&Bs[nxt][lB]  = sB;
        }
#endif
        __syncthreads();
    }
    #pragma unroll
    for (int sm = 0; sm < 2; ++sm) {
        int rowB = mBase + wm * 32 + sm * 16 + ((lane >> 4) << 3);
        #pragma unroll
        for (int sn = 0; sn < 2; ++sn) {
            int n = n0 + wn * 32 + sn * 16 + (lane & 15);
            float bias = dpb[(size_t)e * HD + n];
            #pragma unroll
            for (int r = 0; r < 8; ++r)
                Y[(size_t)(rowB + r) * HD + n] = acc[sm][sn][r] + bias;
        }
    }
}

// ---------------- combine: out[t] = w0*y[row0] + w1*y[row1] ----------------
__global__ __launch_bounds__(256) void k_combine(const float* __restrict__ Y,
                                                 const int* __restrict__ pair_row,
                                                 const float* __restrict__ topk_w,
                                                 float* __restrict__ out) {
    int t = blockIdx.x;
    int r0 = pair_row[t * 2], r1 = pair_row[t * 2 + 1];
    float w0 = topk_w[t * 2], w1 = topk_w[t * 2 + 1];
    const float* y0 = Y + (size_t)r0 * HD;
    const float* y1 = Y + (size_t)r1 * HD;
    float* o = out + (size_t)t * HD;
    for (int h = threadIdx.x; h < HD; h += 256) o[h] = w0 * y0[h] + w1 * y1[h];
}

extern "C" void kernel_launch(void* const* d_in, const int* in_sizes, int n_in,
                              void* d_out, int out_size, void* d_ws, size_t ws_size,
                              hipStream_t stream) {
    (void)in_sizes; (void)n_in; (void)out_size; (void)ws_size;
    const float* x   = (const float*)d_in[0];
    const float* rw  = (const float*)d_in[1];
    const float* rb  = (const float*)d_in[2];
    const float* gup = (const float*)d_in[3];
    const float* gub = (const float*)d_in[4];
    const float* dpw = (const float*)d_in[5];
    const float* dpb = (const float*)d_in[6];
    float* out = (float*)d_out;

    char* base = (char*)d_ws;
    size_t off = 0;
    auto carve = [&](size_t bytes) -> void* {
        void* p = base + off;
        off += (bytes + 255) & ~(size_t)255;
        return p;
    };
    bf16*  Wg       = (bf16*)carve((size_t)NE * ID * HD * 2);
    bf16*  Wu       = (bf16*)carve((size_t)NE * ID * HD * 2);
    bf16*  Wd       = (bf16*)carve((size_t)NE * HD * ID * 2);
    bf16*  Xs       = (bf16*)carve((size_t)PADROWS * HD * 2);
    bf16*  Ac       = (bf16*)carve((size_t)PADROWS * ID * 2);
    float* Y        = (float*)carve((size_t)PADROWS * HD * 4);
    int*   topk_i   = (int*)carve((size_t)NT * 2 * 4);
    float* topk_w   = (float*)carve((size_t)NT * 2 * 4);
    int*   counts   = (int*)carve(NE * 4);
    int*   cursor   = (int*)carve(NE * 4);
    int*   rowBase  = (int*)carve(NE * 4);
    int*   meta     = (int*)carve(4);
    int*   tileE    = (int*)carve(MAXTILES * 4);
    int*   tileRow  = (int*)carve(MAXTILES * 4);
    int*   pair_row = (int*)carve((size_t)NT * 2 * 4);

    k_init<<<1, 32, 0, stream>>>(counts, cursor);
    k_conv_gateup<<<NE * 45 * 45, 256, 0, stream>>>(gup, Wg, Wu);
    k_conv_down<<<NE * 45 * 45, 256, 0, stream>>>(dpw, Wd);
    k_router<<<NT / 8, 256, 0, stream>>>(x, rw, rb, topk_i, topk_w, counts);
    k_tiles<<<1, 1, 0, stream>>>(counts, rowBase, meta, tileE, tileRow);
    k_scatter<<<NT * 2, 256, 0, stream>>>(x, topk_i, rowBase, cursor, pair_row, Xs);
    k_gemm1<<<dim3(ID / BN, MAXTILES), 256, 0, stream>>>(Xs, Wg, Wu, gub, Ac, meta, tileE, tileRow);
    k_gemm2<<<dim3(HD / BN, MAXTILES), 256, 0, stream>>>(Ac, Wd, dpb, Y, meta, tileE, tileRow);
    k_combine<<<NT, 256, 0, stream>>>(Y, pair_row, topk_w, out);
}
